// Attention_69861938037190
// MI455X (gfx1250) — matrix operand order repaired and checked by reading, not compiled, not run on hardware
//
#include <hip/hip_runtime.h>

// ---------------------------------------------------------------------------
// Attention block for MI455X (gfx1250, wave32, WMMA bf16 16x16x32).
//   x:(2,2048,4096) f32, wq:(4096,4096), wk/wv:(1024,4096), wo:(4096,4096)
//   out:(2,2048,4096) f32
// Pipeline: f32->bf16 convert -> Q/K GEMM+RoPE, V GEMM (transposed) ->
//           flash attention (async double-buffered K/V in LDS) -> O GEMM.
// ---------------------------------------------------------------------------

#define DIMK   4096
#define SEQ    2048
#define NH     32
#define NKV    8
#define HD     128
#define BATCH  2
#define TOK    (BATCH * SEQ)            // 4096 tokens
#define SCALE  0.08838834764831845f     // 1/sqrt(128)

typedef __attribute__((ext_vector_type(16))) __bf16 v16bf;
typedef __attribute__((ext_vector_type(8)))  float  v8f;

union FragB { v16bf v; uint4 u[2]; };

__device__ __forceinline__ v8f wmma_bf16(v16bf a, v16bf b, v8f c) {
  // (neg_a, A, neg_b, B, c_mod, C, reuse_a, reuse_b)
  return __builtin_amdgcn_wmma_f32_16x16x32_bf16(false, a, false, b, (short)0, c,
                                                 false, false);
}

// A fragment (16x32 bf16, row-major source). ISA 7.12.2:
//  lane L: m = L&15, half = L>>4; elems 0..7 -> K = half*8+e,
//  elems 8..15 -> K = 16 + half*8 + (e-8).  Two contiguous 16B loads.
__device__ __forceinline__ v16bf load_a_frag(const __bf16* A, int lda, int row0,
                                             int k0, int lane) {
  const int m  = lane & 15;
  const int hf = lane >> 4;
  const __bf16* p = A + (size_t)(row0 + m) * lda + k0 + hf * 8;
  FragB f;
  f.u[0] = *(const uint4*)(p);
  f.u[1] = *(const uint4*)(p + 16);
  return f.v;
}

// B fragment (K=32 x N=16). Source holds N rows of K columns (i.e. B^T
// row-major, e.g. a weight matrix stored [out_feature][in_feature]).
//  lane L: n = L&15, half = L>>4; elems 0..7 -> K = half*8+e,
//  elems 8..15 -> K = 16 + half*8 + (e-8).  Two contiguous 16B loads.
__device__ __forceinline__ v16bf load_bt_frag(const __bf16* Bt, int ldb, int k0,
                                              int lane) {
  const int n  = lane & 15;
  const int hf = lane >> 4;
  const __bf16* p = Bt + (size_t)n * ldb + k0 + hf * 8;
  FragB f;
  f.u[0] = *(const uint4*)(p);
  f.u[1] = *(const uint4*)(p + 16);
  return f.v;
}

// Async global->LDS b128 (ASYNCcnt). INST_OFFSET is added to both the global
// address and the LDS address (ISA 08 §4.4), so one base pair covers 32B.
__device__ __forceinline__ void async_copy_32B(unsigned lds_off,
                                               unsigned long long gaddr) {
  asm volatile("global_load_async_to_lds_b128 %0, %1, off"
               :: "v"(lds_off), "v"(gaddr) : "memory");
  asm volatile("global_load_async_to_lds_b128 %0, %1, off offset:16"
               :: "v"(lds_off), "v"(gaddr) : "memory");
}

// ---------------------------------------------------------------------------
// f32 -> bf16 conversion, 8 elements per thread
// ---------------------------------------------------------------------------
__global__ __launch_bounds__(256) void cvt_f32_bf16(const float* __restrict__ in,
                                                    __bf16* __restrict__ out,
                                                    int n) {
  int i = (blockIdx.x * blockDim.x + threadIdx.x) * 8;
  if (i + 7 < n) {
    union { __bf16 h[8]; uint4 u; } ob;
#pragma unroll
    for (int e = 0; e < 8; ++e) ob.h[e] = (__bf16)in[i + e];
    *(uint4*)(out + i) = ob.u;
  } else {
    for (int e = i; e < n; ++e) out[e] = (__bf16)in[e];
  }
}

// ---------------------------------------------------------------------------
// Projection GEMM. A: TOK x 4096 bf16, W: N x 4096 bf16 (row-major = B^T).
// Each wave computes a 32-row x 64-col tile: 2 M-subtiles x 4 N-tiles,
// 8 WMMAs per K-step for 12 b128 loads (1.5 loads/WMMA).
// MODE 0: Q  cols {c0,c0+16,c0+64,c0+80} of head hh; RoPE; out [b][h][s][d]
// MODE 1: K  same, kv heads
// MODE 2: V  no RoPE; out transposed [b][kvh][d][s]
// MODE 3: O  cols {c0..c0+48}; f32 out [t][4096]
// ---------------------------------------------------------------------------
template <int MODE>
__global__ __launch_bounds__(256) void gemm_proj(const __bf16* __restrict__ A,
                                                 const __bf16* __restrict__ W,
                                                 const float* __restrict__ cosp,
                                                 const float* __restrict__ sinp,
                                                 __bf16* __restrict__ outb,
                                                 float* __restrict__ outf) {
  const int lane = threadIdx.x & 31;
  const int wave = threadIdx.x >> 5;
  const int mt   = blockIdx.x * 8 + wave;      // 0..127
  const int row0 = mt * 32;

  int cb[4];
  int hh = 0, dt2 = 0;
  if (MODE == 3) {
    const int c0 = blockIdx.y * 64;
    cb[0] = c0; cb[1] = c0 + 16; cb[2] = c0 + 32; cb[3] = c0 + 48;
  } else {
    hh  = blockIdx.y >> 1;                     // head
    dt2 = blockIdx.y & 1;                      // 32-col slab of first half
    const int c0 = hh * HD + dt2 * 32;         // d in [0,64)
    cb[0] = c0; cb[1] = c0 + 16;               // rotate-half partners:
    cb[2] = c0 + 64; cb[3] = c0 + 80;          //   cb[0]<->cb[2], cb[1]<->cb[3]
  }

  const __bf16* wp[4];
#pragma unroll
  for (int c = 0; c < 4; ++c) wp[c] = W + (size_t)cb[c] * DIMK;

  const v8f z8 = {0, 0, 0, 0, 0, 0, 0, 0};
  v8f acc[2][4];
#pragma unroll
  for (int mi = 0; mi < 2; ++mi)
#pragma unroll
    for (int c = 0; c < 4; ++c) acc[mi][c] = z8;

  for (int k0 = 0; k0 < DIMK; k0 += 32) {
    const v16bf a0 = load_a_frag(A, DIMK, row0,      k0, lane);
    const v16bf a1 = load_a_frag(A, DIMK, row0 + 16, k0, lane);
    __builtin_prefetch(A + (size_t)(row0 + (lane & 15)) * DIMK + k0 + 256, 0, 1);
#pragma unroll
    for (int c = 0; c < 4; ++c) {
      const v16bf b = load_bt_frag(wp[c], DIMK, k0, lane);
      acc[0][c] = wmma_bf16(a0, b, acc[0][c]);
      acc[1][c] = wmma_bf16(a1, b, acc[1][c]);
    }
  }

  const int n  = lane & 15;
  const int hf = lane >> 4;

  if (MODE == 0 || MODE == 1) {
#pragma unroll
    for (int mi = 0; mi < 2; ++mi)
#pragma unroll
      for (int pj = 0; pj < 2; ++pj)
#pragma unroll
        for (int j = 0; j < 8; ++j) {
          const int t  = row0 + mi * 16 + j + 8 * hf;
          const int s  = t & (SEQ - 1);
          const int bb = t >> 11;
          const int d  = dt2 * 32 + pj * 16 + n;     // < 64
          const float c  = cosp[s * HD + d];
          const float si = sinp[s * HD + d];
          const float a0 = acc[mi][pj][j];           // col d
          const float a1 = acc[mi][pj + 2][j];       // col d+64
          const size_t base = (MODE == 0)
              ? (((size_t)bb * NH  + hh) * SEQ + s) * HD
              : (((size_t)bb * NKV + hh) * SEQ + s) * HD;
          outb[base + d]      = (__bf16)(a0 * c - a1 * si);
          outb[base + d + 64] = (__bf16)(a1 * c + a0 * si);
        }
  } else if (MODE == 2) {
#pragma unroll
    for (int mi = 0; mi < 2; ++mi)
#pragma unroll
      for (int pj = 0; pj < 2; ++pj)
#pragma unroll
        for (int j = 0; j < 8; ++j) {
          const int t  = row0 + mi * 16 + j + 8 * hf;
          const int s  = t & (SEQ - 1);
          const int bb = t >> 11;
          const int d  = dt2 * 32 + pj * 16 + n;
          const size_t base = ((size_t)bb * NKV + hh) * HD;
          outb[(base + d)      * SEQ + s] = (__bf16)acc[mi][pj][j];
          outb[(base + d + 64) * SEQ + s] = (__bf16)acc[mi][pj + 2][j];
        }
  } else {
#pragma unroll
    for (int mi = 0; mi < 2; ++mi)
#pragma unroll
      for (int c = 0; c < 4; ++c)
#pragma unroll
        for (int j = 0; j < 8; ++j) {
          const int t = row0 + mi * 16 + j + 8 * hf;
          outf[(size_t)t * DIMK + cb[c] + n] = acc[mi][c][j];
        }
  }
}

// ---------------------------------------------------------------------------
// Flash attention. Block = 8 waves = 8 consecutive 16-query tiles of one
// (batch, head); all waves share the same K/V head. K (32x128) and V^T
// (128x32) tiles are double-buffered in LDS via async global->LDS copies
// (4 b128 per thread per tile; ASYNCcnt completes in order so a counted
// s_wait_asynccnt 0x4 releases the current tile while the prefetch of the
// next one is still in flight).
// q:[b][h][s][128], k:[b][kvh][s][128], vt:[b][kvh][128][s] bf16 -> ao:[t][4096]
// ---------------------------------------------------------------------------
__global__ __launch_bounds__(256) void attn_kernel(const __bf16* __restrict__ q,
                                                   const __bf16* __restrict__ k,
                                                   const __bf16* __restrict__ vt,
                                                   __bf16* __restrict__ ao) {
  __shared__ alignas(16) __bf16 kts[2][32][HD];    // 16 KB: keys x dims
  __shared__ alignas(16) __bf16 vts[2][HD][32];    // 16 KB: dims x keys
  __shared__ alignas(16) __bf16 stage[8][16][32];  //  8 KB: per-wave P

  const int tid  = threadIdx.x;
  const int lane = tid & 31;
  const int wave = tid >> 5;
  const int qt   = blockIdx.x * 8 + wave;      // 0..127
  const int h    = blockIdx.y;                 // 0..31
  const int b    = blockIdx.z;                 // 0..1
  const int kvh  = h >> 2;                     // 4:1 GQA
  const int n    = lane & 15;
  const int hf   = lane >> 4;

  const __bf16* qb = q  + (((size_t)b * NH  + h)   * SEQ) * HD;
  const __bf16* kb = k  + (((size_t)b * NKV + kvh) * SEQ) * HD;
  const __bf16* vb = vt + (((size_t)b * NKV + kvh) * HD)  * SEQ;

  v16bf aq[4];
#pragma unroll
  for (int dc = 0; dc < 4; ++dc)
    aq[dc] = load_a_frag(qb, HD, qt * 16, dc * 32, lane);

  const v8f z8 = {0, 0, 0, 0, 0, 0, 0, 0};
  v8f acc[8];
  float mi[8], li[8];
#pragma unroll
  for (int i = 0; i < 8; ++i) { acc[i] = z8; mi[i] = -3.0e38f; li[i] = 0.f; }

  // cooperative staging slots (32B per thread per tensor per tile)
  const int krow = tid >> 3;            // 0..31  (key within tile)
  const int kcol = (tid & 7) * 16;      // element col in [0,128)
  const int vrow = tid >> 1;            // 0..127 (dim)
  const int vcol = (tid & 1) * 16;      // element col in [0,32)

  auto issue_tile = [&](int kt, int bufi) {
    async_copy_32B((unsigned)(size_t)&kts[bufi][krow][kcol],
                   (unsigned long long)(size_t)(kb + (size_t)(kt * 32 + krow) * HD + kcol));
    async_copy_32B((unsigned)(size_t)&vts[bufi][vrow][vcol],
                   (unsigned long long)(size_t)(vb + (size_t)vrow * SEQ + kt * 32 + vcol));
  };

  const int qt_max  = blockIdx.x * 8 + 7;                 // last wave's tile
  const int nkt     = ((qt_max * 16 + 15) + 32) >> 5;     // block-uniform trips
  const int own_nkt = ((qt     * 16 + 15) + 32) >> 5;

  issue_tile(0, 0);
  for (int kt = 0; kt < nkt; ++kt) {
    const int cur  = kt & 1;
    const bool more = (kt + 1 < nkt);
    if (more) {
      issue_tile(kt + 1, cur ^ 1);
      asm volatile("s_wait_asynccnt 0x4" ::: "memory");   // current tile done
    } else {
      asm volatile("s_wait_asynccnt 0x0" ::: "memory");
    }
    __syncthreads();                                      // publish across waves

    if (kt < own_nkt) {
      const int kbase = kt * 32;

      // S = Q (16x128) x K^T (128x32), K fragments from LDS
      v8f s0 = z8, s1 = z8;
#pragma unroll
      for (int dc = 0; dc < 4; ++dc) {
        const v16bf bk0 = load_bt_frag(&kts[cur][0][0],  HD, dc * 32, lane);
        const v16bf bk1 = load_bt_frag(&kts[cur][16][0], HD, dc * 32, lane);
        s0 = wmma_bf16(aq[dc], bk0, s0);
        s1 = wmma_bf16(aq[dc], bk1, s1);
      }

      // scale + causal mask + online softmax (rows live in 16-lane halves)
#pragma unroll
      for (int j = 0; j < 8; ++j) {
        const int qrow = qt * 16 + j + 8 * hf;
        float v0 = s0[j] * SCALE;
        float v1 = s1[j] * SCALE;
        if (kbase + n      > qrow) v0 = -3.0e38f;
        if (kbase + 16 + n > qrow) v1 = -3.0e38f;

        float mx = fmaxf(v0, v1);
#pragma unroll
        for (int msk = 1; msk < 16; msk <<= 1)
          mx = fmaxf(mx, __shfl_xor(mx, msk, 32));
        const float mnew  = fmaxf(mi[j], mx);
        const float alpha = __expf(mi[j] - mnew);
        const float p0 = __expf(v0 - mnew);
        const float p1 = __expf(v1 - mnew);
        float rs = p0 + p1;
#pragma unroll
        for (int msk = 1; msk < 16; msk <<= 1)
          rs += __shfl_xor(rs, msk, 32);
        li[j] = li[j] * alpha + rs;
        mi[j] = mnew;
#pragma unroll
        for (int nt = 0; nt < 8; ++nt) acc[nt][j] *= alpha;

        const int prow = j + 8 * hf;
        stage[wave][prow][n]      = (__bf16)p0;
        stage[wave][prow][16 + n] = (__bf16)p1;
      }

      // wave-private LDS round trip: C-layout -> A-fragment layout
      asm volatile("s_wait_dscnt 0" ::: "memory");
      FragB pf;
      const __bf16* sp = &stage[wave][n][0];
      pf.u[0] = *(const uint4*)(sp + hf * 8);
      pf.u[1] = *(const uint4*)(sp + 16 + hf * 8);

      // O += P (16x32) x V (32x128), V fragments from LDS (V^T rows)
#pragma unroll
      for (int nt = 0; nt < 8; ++nt) {
        const v16bf bv = load_bt_frag(&vts[cur][nt * 16][0], 32, 0, lane);
        acc[nt] = wmma_bf16(pf.v, bv, acc[nt]);
      }
    }

    __syncthreads();   // all waves done reading buffers before re-issue
  }

  // epilogue: normalize and store bf16 [t][h*128 + d]
#pragma unroll
  for (int j = 0; j < 8; ++j) {
    const float inv = 1.0f / li[j];
    const size_t t  = (size_t)b * SEQ + qt * 16 + j + 8 * hf;
#pragma unroll
    for (int nt = 0; nt < 8; ++nt)
      ao[t * DIMK + h * HD + nt * 16 + n] = (__bf16)(acc[nt][j] * inv);
  }
}

// ---------------------------------------------------------------------------
// Workspace layout (bytes). Total = 201,326,592 (~192 MiB).
// ---------------------------------------------------------------------------
#define OFF_XB   0                         // x bf16        33554432
#define OFF_WQB  33554432                  // wq bf16       33554432
#define OFF_WKB  67108864                  // wk bf16        8388608
#define OFF_WVB  75497472                  // wv bf16        8388608
#define OFF_WOB  83886080                  // wo bf16       33554432
#define OFF_Q    117440512                 // q  bf16       33554432
#define OFF_K    150994944                 // k  bf16        8388608
#define OFF_VT   159383552                 // v^T bf16       8388608
#define OFF_AO   167772160                 // attn out bf16 33554432

extern "C" void kernel_launch(void* const* d_in, const int* in_sizes, int n_in,
                              void* d_out, int out_size, void* d_ws, size_t ws_size,
                              hipStream_t stream) {
  const float* x    = (const float*)d_in[0];
  const float* cosp = (const float*)d_in[1];
  const float* sinp = (const float*)d_in[2];
  const float* wq   = (const float*)d_in[3];
  const float* wk   = (const float*)d_in[4];
  const float* wv   = (const float*)d_in[5];
  const float* wo   = (const float*)d_in[6];
  float* out = (float*)d_out;

  char* ws = (char*)d_ws;
  __bf16* xb  = (__bf16*)(ws + OFF_XB);
  __bf16* wqb = (__bf16*)(ws + OFF_WQB);
  __bf16* wkb = (__bf16*)(ws + OFF_WKB);
  __bf16* wvb = (__bf16*)(ws + OFF_WVB);
  __bf16* wob = (__bf16*)(ws + OFF_WOB);
  __bf16* qb  = (__bf16*)(ws + OFF_Q);
  __bf16* kk  = (__bf16*)(ws + OFF_K);
  __bf16* vtb = (__bf16*)(ws + OFF_VT);
  __bf16* aob = (__bf16*)(ws + OFF_AO);

  const int nx = TOK * DIMK;        // 16777216
  const int nw = DIMK * DIMK;       // 16777216
  const int nk = NKV * HD * DIMK;   //  4194304
  dim3 blk(256);
  cvt_f32_bf16<<<dim3((nx / 8 + 255) / 256), blk, 0, stream>>>(x,  xb,  nx);
  cvt_f32_bf16<<<dim3((nw / 8 + 255) / 256), blk, 0, stream>>>(wq, wqb, nw);
  cvt_f32_bf16<<<dim3((nk / 8 + 255) / 256), blk, 0, stream>>>(wk, wkb, nk);
  cvt_f32_bf16<<<dim3((nk / 8 + 255) / 256), blk, 0, stream>>>(wv, wvb, nk);
  cvt_f32_bf16<<<dim3((nw / 8 + 255) / 256), blk, 0, stream>>>(wo, wob, nw);

  // 128 m-tiles (32 rows each) / 8 waves per block -> grid.x = 16
  gemm_proj<0><<<dim3(16, NH  * 2), blk, 0, stream>>>(xb, wqb, cosp, sinp, qb,  nullptr);
  gemm_proj<1><<<dim3(16, NKV * 2), blk, 0, stream>>>(xb, wkb, cosp, sinp, kk,  nullptr);
  gemm_proj<2><<<dim3(16, NKV * 2), blk, 0, stream>>>(xb, wvb, nullptr, nullptr, vtb, nullptr);

  attn_kernel<<<dim3(16, NH, BATCH), blk, 0, stream>>>(qb, kk, vtb, aob);

  gemm_proj<3><<<dim3(16, DIMK / 64), blk, 0, stream>>>(aob, wob, nullptr, nullptr,
                                                        nullptr, out);
}
